// MambaBlock_18253611008296
// MI455X (gfx1250) — matrix operand order repaired and checked by reading, not compiled, not run on hardware
//
#include <hip/hip_runtime.h>

// ---------------------------------------------------------------------------
// Mamba block forward for MI455X (gfx1250, wave32, WMMA).
// GEMMs: bf16 inputs, fp32 accumulation via v_wmma_f32_16x16x32_bf16.
// Scan: async global->LDS double-buffered B/C staging (ASYNCcnt) when available.
// ---------------------------------------------------------------------------

typedef __attribute__((ext_vector_type(16))) __bf16 v16bf;
typedef __attribute__((ext_vector_type(8)))  __bf16 v8bf;
typedef __attribute__((ext_vector_type(8)))  float  v8f;
typedef __attribute__((ext_vector_type(4)))  int    v4i;

typedef __attribute__((address_space(1))) v4i* as1_v4i_ptr;
typedef __attribute__((address_space(3))) v4i* as3_v4i_ptr;

#define BSZ   2
#define SEQL  2048
#define DM    1024     // d_model
#define DI    2048     // d_inner
#define NTOK  (BSZ*SEQL)   // 4096 token rows
#define NST   16       // d_state
#define DTR   64       // dt_rank
#define XC    (DTR + 2*NST)  // 96

#if __has_builtin(__builtin_amdgcn_global_load_async_to_lds_b128) && \
    __has_builtin(__builtin_amdgcn_s_wait_asynccnt)
#define HAS_ASYNC_LDS 1
#else
#define HAS_ASYNC_LDS 0
#endif

// ------------------------------ fp32 -> bf16 -------------------------------
__global__ __launch_bounds__(256)
void cvt_bf16_kernel(const float* __restrict__ src, __bf16* __restrict__ dst, int n)
{
    int i = blockIdx.x * 256 + threadIdx.x;
    if (i < n) dst[i] = (__bf16)src[i];
}

// ------------------------------ LayerNorm ----------------------------------
// One block (256 threads, 8 waves) per token row of 1024.
__global__ __launch_bounds__(256)
void ln_kernel(const float* __restrict__ x, const float* __restrict__ w,
               const float* __restrict__ b, __bf16* __restrict__ out)
{
    const int row = blockIdx.x;
    const float* xr = x + (size_t)row * DM;

    float s = 0.f, s2 = 0.f;
    #pragma unroll
    for (int i = threadIdx.x; i < DM; i += 256) {
        float v = xr[i];
        s += v; s2 += v * v;
    }
    #pragma unroll
    for (int off = 16; off; off >>= 1) {
        s  += __shfl_xor(s,  off);
        s2 += __shfl_xor(s2, off);
    }
    __shared__ float red[16];
    const int lane = threadIdx.x & 31, wid = threadIdx.x >> 5;
    if (lane == 0) { red[wid] = s; red[8 + wid] = s2; }
    __syncthreads();
    if (threadIdx.x == 0) {
        float ts = 0.f, ts2 = 0.f;
        #pragma unroll
        for (int i = 0; i < 8; i++) { ts += red[i]; ts2 += red[8 + i]; }
        red[0] = ts; red[1] = ts2;
    }
    __syncthreads();
    const float mu  = red[0] * (1.f / DM);
    const float var = red[1] * (1.f / DM) - mu * mu;
    const float inv = rsqrtf(var + 1e-5f);

    __bf16* orow = out + (size_t)row * DM;
    #pragma unroll
    for (int i = threadIdx.x; i < DM; i += 256)
        orow[i] = (__bf16)((xr[i] - mu) * inv * w[i] + b[i]);
}

// --------------------- WMMA bf16 GEMM: C = A(M,K) * W(N,K)^T ---------------
// Each wave computes a (16*MSUB) x (16*NSUB) strip; MSUB*NSUB independent
// accumulators keep the WMMA pipeline full (bf16 RAW needs 5 slots).
// Requires: M % (16*MSUB) == 0, N % (16*NSUB) == 0, K % 32 == 0.
// EPI 0: plain store.  EPI 1: softplus(acc + bias[n]).  EPI 2: acc + res[m,n].
template <int EPI, int MSUB, int NSUB>
__global__ __launch_bounds__(256)
void gemm_bf16_kernel(const __bf16* __restrict__ A, const __bf16* __restrict__ W,
                      float* __restrict__ C, int M, int N, int K,
                      const float* __restrict__ extra)
{
    const int lane   = threadIdx.x & 31;
    const int wave   = blockIdx.x * 8 + (threadIdx.x >> 5);
    const int tilesM = M / (16 * MSUB);
    const int tilesN = N / (16 * NSUB);
    const int tileM  = wave / tilesN;
    const int tileN  = wave - tileM * tilesN;
    if (tileM >= tilesM) return;           // wave-uniform tail guard

    const int laneHi = lane >> 4;          // 0|1
    const int lane16 = lane & 15;

    // A lane layout (16x32 bf16): lane<16 -> K k..k+15 contiguous of its row;
    // lane>=16 -> K k+16..k+31.
    const __bf16* pA[MSUB];
    #pragma unroll
    for (int mi = 0; mi < MSUB; mi++)
        pA[mi] = A + (size_t)(tileM * MSUB * 16 + mi * 16 + lane16) * K + laneHi * 16;

    // B lane layout (32x16 bf16): lane n<16 -> column n, K k..k+15 contiguous;
    // lane>=16 -> column n-16, K k+16..k+31.
    const __bf16* pW[NSUB];
    #pragma unroll
    for (int s = 0; s < NSUB; s++)
        pW[s] = W + (size_t)(tileN * NSUB * 16 + s * 16 + lane16) * K + laneHi * 16;

    v8f acc[MSUB][NSUB] = {};

    for (int k = 0; k < K; k += 32) {
        v16bf a[MSUB];
        #pragma unroll
        for (int mi = 0; mi < MSUB; mi++) {
            __builtin_prefetch((const void*)(pA[mi] + k + 256), 0, 3);
            v8bf a0 = *(const v8bf*)(pA[mi] + k);
            v8bf a1 = *(const v8bf*)(pA[mi] + k + 8);
            #pragma unroll
            for (int i = 0; i < 8; i++) { a[mi][i] = a0[i]; a[mi][i + 8] = a1[i]; }
        }
        #pragma unroll
        for (int s = 0; s < NSUB; s++) {
            v16bf bm = *(const v16bf*)(pW[s] + k);
            #pragma unroll
            for (int mi = 0; mi < MSUB; mi++) {
                acc[mi][s] = __builtin_amdgcn_wmma_f32_16x16x32_bf16(
                    false, a[mi], false, bm, (short)0, acc[mi][s], false, false);
            }
        }
    }

    // C/D layout: VGPR r -> row (laneHi*8 + r), col lane16.
    #pragma unroll
    for (int mi = 0; mi < MSUB; mi++) {
        #pragma unroll
        for (int s = 0; s < NSUB; s++) {
            const int n = tileN * NSUB * 16 + s * 16 + lane16;
            #pragma unroll
            for (int r = 0; r < 8; r++) {
                const int m = tileM * MSUB * 16 + mi * 16 + laneHi * 8 + r;
                float v = acc[mi][s][r];
                if (EPI == 1) {
                    float t = v + extra[n];
                    v = (t > 20.f) ? t : log1pf(__expf(t));   // softplus
                } else if (EPI == 2) {
                    v += extra[(size_t)m * N + n];            // residual
                }
                C[(size_t)m * N + n] = v;
            }
        }
    }
}

// --------------- causal depthwise conv1d (w=4) + bias + SiLU ---------------
// u comes from columns [0,DI) of xz (NTOK x 2*DI). Writes fp32 + bf16 copies.
__global__ __launch_bounds__(256)
void conv_silu_kernel(const float* __restrict__ xz, const float* __restrict__ cw,
                      const float* __restrict__ cb, float* __restrict__ u,
                      __bf16* __restrict__ u_bf)
{
    int idx = blockIdx.x * 256 + threadIdx.x;       // over NTOK*DI
    if (idx >= NTOK * DI) return;
    const int d = idx & (DI - 1);
    const int t = idx >> 11;                        // token row (DI = 2048)
    const int l = t & (SEQL - 1);

    float acc = cb[d];
    #pragma unroll
    for (int j = 0; j < 4; j++) {
        int ll = l - 3 + j;
        if (ll >= 0)
            acc += cw[d * 4 + j] * xz[(size_t)(t - 3 + j) * (2 * DI) + d];
    }
    float sv = acc / (1.f + __expf(-acc));          // SiLU
    u[idx]    = sv;
    u_bf[idx] = (__bf16)sv;
}

// ---------------- extract dt (first 64 cols of x_dbl) as bf16 --------------
__global__ __launch_bounds__(256)
void dt_extract_kernel(const float* __restrict__ xdbl, __bf16* __restrict__ dtb)
{
    int i = blockIdx.x * 256 + threadIdx.x;         // over NTOK*DTR
    if (i >= NTOK * DTR) return;
    const int row = i >> 6, col = i & 63;
    dtb[i] = (__bf16)xdbl[(size_t)row * XC + col];
}

// ------------------------------ selective scan -----------------------------
// 64 threads/block, one channel d per thread, 16-state recurrence in regs.
// B/C (cols 64..95 of x_dbl, 128B per row) streamed into LDS with async
// global->LDS b128 copies (ASYNCcnt), double-buffered across 8-step chunks.
__global__ __launch_bounds__(64)
void scan_kernel(const float* __restrict__ delta, const float* __restrict__ xdbl,
                 const float* __restrict__ u, const float* __restrict__ xz,
                 const float* __restrict__ A_log, const float* __restrict__ Dp,
                 __bf16* __restrict__ y_bf)
{
    const int b = blockIdx.x >> 5;                  // 32 d-chunks per batch
    const int d = ((blockIdx.x & 31) << 6) + threadIdx.x;

    float a[NST], h[NST];
    #pragma unroll
    for (int n = 0; n < NST; n++) {
        a[n] = -__expf(A_log[(size_t)d * NST + n]);
        h[n] = 0.f;
    }
    const float Dd = Dp[d];

    // [buf][row-in-chunk][ B[0..15] | C[16..31] ]
    __shared__ __align__(16) float sBC[2][8][32];

    const int rsel = threadIdx.x >> 3;              // row in chunk (0..7)
    const int psel = threadIdx.x & 7;               // 16B segment (0..7)

    auto stage = [&](int l0, int buf) {
        const float* g = xdbl + (size_t)(b * SEQL + l0 + rsel) * XC + DTR + psel * 4;
        float* l = &sBC[buf][rsel][psel * 4];
#if HAS_ASYNC_LDS
        __builtin_amdgcn_global_load_async_to_lds_b128(
            (as1_v4i_ptr)(v4i*)const_cast<float*>(g),
            (as3_v4i_ptr)(v4i*)l, 0, 0);
#else
        *(float4*)l = *(const float4*)g;
#endif
    };

    constexpr int NCHUNK = SEQL / 8;
    stage(0, 0);
    for (int c = 0; c < NCHUNK; c++) {
        const int cur = c & 1;
        if (c + 1 < NCHUNK) {
            stage((c + 1) * 8, cur ^ 1);
#if HAS_ASYNC_LDS
            __builtin_amdgcn_s_wait_asynccnt(1);    // chunk c complete (in-order)
#endif
        } else {
#if HAS_ASYNC_LDS
            __builtin_amdgcn_s_wait_asynccnt(0);
#endif
        }
        __syncthreads();                            // all lanes' stages visible

        #pragma unroll
        for (int lt = 0; lt < 8; lt++) {
            const size_t row = (size_t)(b * SEQL + c * 8 + lt);
            const size_t tix = row * DI + d;
            const float dl = delta[tix];
            const float uu = u[tix];
            const float zz = xz[row * (2 * DI) + DI + d];
            const float du = dl * uu;
            float y = 0.f;
            #pragma unroll
            for (int n = 0; n < NST; n++) {
                const float dA = __expf(dl * a[n]);
                h[n] = dA * h[n] + du * sBC[cur][lt][n];
                y += h[n] * sBC[cur][lt][16 + n];
            }
            y += uu * Dd;
            const float sz = zz / (1.f + __expf(-zz));
            y_bf[tix] = (__bf16)(y * sz);
        }
        __syncthreads();                            // done reading buf `cur`
    }
}

// ---------------------------------------------------------------------------
extern "C" void kernel_launch(void* const* d_in, const int* in_sizes, int n_in,
                              void* d_out, int out_size, void* d_ws, size_t ws_size,
                              hipStream_t stream)
{
    (void)in_sizes; (void)n_in; (void)out_size; (void)ws_size;

    const float* x       = (const float*)d_in[0];   // (B,L,DM)
    const float* ln_w    = (const float*)d_in[1];
    const float* ln_b    = (const float*)d_in[2];
    const float* in_proj = (const float*)d_in[3];   // (2*DI, DM)
    const float* conv_w  = (const float*)d_in[4];   // (DI, 4)
    const float* conv_b  = (const float*)d_in[5];
    const float* x_proj  = (const float*)d_in[6];   // (XC, DI)
    const float* dt_w    = (const float*)d_in[7];   // (DI, DTR)
    const float* dt_b    = (const float*)d_in[8];
    const float* A_log   = (const float*)d_in[9];   // (DI, NST)
    const float* Dp      = (const float*)d_in[10];
    const float* out_w   = (const float*)d_in[11];  // (DM, DI)
    float* out = (float*)d_out;                     // (B,L,DM)

    // ---- workspace carve-out (256B aligned) ----
    char* p = (char*)d_ws;
    auto carve = [&](size_t bytes) -> char* {
        char* r = p; p += (bytes + 255) & ~(size_t)255; return r;
    };
    __bf16* wInBf  = (__bf16*)carve((size_t)2 * DI * DM * 2);
    __bf16* wXBf   = (__bf16*)carve((size_t)XC * DI * 2);
    __bf16* wDtBf  = (__bf16*)carve((size_t)DI * DTR * 2);
    __bf16* wOutBf = (__bf16*)carve((size_t)DM * DI * 2);
    __bf16* xnBf   = (__bf16*)carve((size_t)NTOK * DM * 2);
    float*  xz     = (float*) carve((size_t)NTOK * 2 * DI * 4);
    float*  u      = (float*) carve((size_t)NTOK * DI * 4);
    __bf16* uBf    = (__bf16*)carve((size_t)NTOK * DI * 2);
    float*  xdbl   = (float*) carve((size_t)NTOK * XC * 4);
    __bf16* dtBf   = (__bf16*)carve((size_t)NTOK * DTR * 2);
    float*  dlt    = (float*) carve((size_t)NTOK * DI * 4);
    __bf16* yBf    = (__bf16*)carve((size_t)NTOK * DI * 2);

    auto gemm_blocks = [](int M, int N, int msub, int nsub) {
        int waves = (M / (16 * msub)) * (N / (16 * nsub));
        return (waves + 7) / 8;
    };

    // 1) weights -> bf16
    {
        int n1 = 2 * DI * DM, n2 = XC * DI, n3 = DI * DTR, n4 = DM * DI;
        cvt_bf16_kernel<<<(n1 + 255) / 256, 256, 0, stream>>>(in_proj, wInBf, n1);
        cvt_bf16_kernel<<<(n2 + 255) / 256, 256, 0, stream>>>(x_proj,  wXBf,  n2);
        cvt_bf16_kernel<<<(n3 + 255) / 256, 256, 0, stream>>>(dt_w,    wDtBf, n3);
        cvt_bf16_kernel<<<(n4 + 255) / 256, 256, 0, stream>>>(out_w,   wOutBf, n4);
    }

    // 2) LayerNorm -> xnBf
    ln_kernel<<<NTOK, 256, 0, stream>>>(x, ln_w, ln_b, xnBf);

    // 3) in_proj: xz = xn @ in_proj^T   (M=4096, N=4096, K=1024)
    gemm_bf16_kernel<0, 2, 4><<<gemm_blocks(NTOK, 2 * DI, 2, 4), 256, 0, stream>>>(
        xnBf, wInBf, xz, NTOK, 2 * DI, DM, nullptr);

    // 4) causal depthwise conv + SiLU -> u (fp32 + bf16)
    conv_silu_kernel<<<(NTOK * DI + 255) / 256, 256, 0, stream>>>(
        xz, conv_w, conv_b, u, uBf);

    // 5) x_proj: x_dbl = u @ x_proj^T   (M=4096, N=96, K=2048)  N: 3 x 32-col
    gemm_bf16_kernel<0, 2, 2><<<gemm_blocks(NTOK, XC, 2, 2), 256, 0, stream>>>(
        uBf, wXBf, xdbl, NTOK, XC, DI, nullptr);

    // 6) dt slice -> bf16
    dt_extract_kernel<<<(NTOK * DTR + 255) / 256, 256, 0, stream>>>(xdbl, dtBf);

    // 7) dt_proj + softplus: delta = softplus(dt @ dt_proj^T + dt_bias)
    gemm_bf16_kernel<1, 2, 4><<<gemm_blocks(NTOK, DI, 2, 4), 256, 0, stream>>>(
        dtBf, wDtBf, dlt, NTOK, DI, DTR, dt_b);

    // 8) selective scan + D-skip + SiLU(z) gate -> yBf
    scan_kernel<<<BSZ * (DI / 64), 64, 0, stream>>>(
        dlt, xdbl, u, xz, A_log, Dp, yBf);

    // 9) out_proj + residual: out = y @ out_proj^T + x
    gemm_bf16_kernel<2, 2, 4><<<gemm_blocks(NTOK, DM, 2, 4), 256, 0, stream>>>(
        yBf, wOutBf, out, NTOK, DM, DI, x);
}
